// MultimodalSparseDeformableTransformerDecoderLayer_13872744366703
// MI455X (gfx1250) — matrix operand order repaired and checked by reading, not compiled, not run on hardware
//
#include <hip/hip_runtime.h>
#include <hip/hip_fp16.h>

// ---------------- problem constants ----------------
#define DMODEL 256
#define NHEAD 8
#define DHEAD 32
#define BATCH 16
#define NQ 512
#define STOT 15360
#define BQ (BATCH * NQ)       // 8192
#define BS (BATCH * STOT)     // 245760

typedef __attribute__((ext_vector_type(16))) _Float16 v16h;
typedef __attribute__((ext_vector_type(8)))  float    v8f;

__device__ __forceinline__ v8f wmma_f16(v16h a, v16h b, v8f c) {
    return __builtin_amdgcn_wmma_f32_16x16x32_f16(false, a, false, b, (short)0, c, false, false);
}

// ---- fragment loaders (contiguous k order, same for A and B, wave32) ----
// A (16-bit, 16x32): lane L holds row M=L%16. Element e=2i+j (VGPR i, half j):
//   K = 16*(L>=16) + 2i + j
__device__ __forceinline__ v16h frag_a_lds(const __half* rowbase, int hi) {
    union { v16h v; unsigned u[8]; } r;
#pragma unroll
    for (int i = 0; i < 8; ++i) {
        int k = 16 * hi + 2 * i;
        r.u[i] = *reinterpret_cast<const unsigned*>(rowbase + k);
    }
    return r.v;
}
// B (16-bit, 32x16): lane L holds col N=L%16. Element e=2i+j: K = 16*(L>=16)+2i+j
__device__ __forceinline__ v16h frag_b_lds(const __half* rowbase, int hi) {
    union { v16h v; unsigned u[8]; } r;
#pragma unroll
    for (int i = 0; i < 8; ++i)
        r.u[i] = *reinterpret_cast<const unsigned*>(rowbase + 16 * hi + 2 * i);
    return r.v;
}
__device__ __forceinline__ v16h frag_a_g32(const float* rowbase, int hi) {
    union { v16h v; _Float16 h[16]; } r;
#pragma unroll
    for (int i = 0; i < 8; ++i) {
        int k = 16 * hi + 2 * i;
        r.h[2 * i]     = (_Float16)rowbase[k];
        r.h[2 * i + 1] = (_Float16)rowbase[k + 1];
    }
    return r.v;
}
__device__ __forceinline__ v16h frag_b_row_g32(const float* rowbase, int hi) {
    union { v16h v; _Float16 h[16]; } r;
#pragma unroll
    for (int i = 0; i < 8; ++i) {
        int k = 16 * hi + 2 * i;
        r.h[2 * i]     = (_Float16)rowbase[k];
        r.h[2 * i + 1] = (_Float16)rowbase[k + 1];
    }
    return r.v;
}
__device__ __forceinline__ v16h frag_b_strided_g32(const float* base, int hi, int stride) {
    union { v16h v; _Float16 h[16]; } r;
#pragma unroll
    for (int i = 0; i < 8; ++i) {
        int k = 16 * hi + 2 * i;
        r.h[2 * i]     = (_Float16)base[(size_t)k * stride];
        r.h[2 * i + 1] = (_Float16)base[(size_t)(k + 1) * stride];
    }
    return r.v;
}

__device__ __forceinline__ void store_out(float* p, float v)  { *p = v; }
__device__ __forceinline__ void store_out(__half* p, float v) { *p = __float2half(v); }

// ---------------- generic GEMM: C[M,N] = A[M,K] @ W[N,K]^T + bias ----------------
// Requires M%128==0, N%64==0, K%32==0.
// EPI: 0 = bias, 1 = bias+relu, 2 = bias+rowmask-zero. OutT: float or __half.
// Block tile 128x64, 8 waves as 4(M)x2(N); wave computes 32x32 via 4 WMMAs/K-step.
// Double-buffered LDS, software-pipelined float4 global loads.
template <int EPI, typename OutT>
__global__ __launch_bounds__(256) void gemm_kernel(
    const float* __restrict__ A, const float* __restrict__ W,
    const float* __restrict__ bias, OutT* __restrict__ C,
    int M, int N, int K, const unsigned char* __restrict__ rowmask)
{
    __shared__ __half As[2][128 * 32];
    __shared__ __half Ws[2][64 * 32];
    const int tid = threadIdx.x;
    const int lane = tid & 31, wv = tid >> 5;
    const int wy = wv & 3, wx = wv >> 2;
    const int m0 = blockIdx.y * 128, n0 = blockIdx.x * 64;
    const int hi = lane >> 4, l16 = lane & 15;

    // staging: each thread owns 4 A-rows (float4) and 2 W-rows (float4)
    const int sr = tid >> 3, sc = (tid & 7) * 4;
    float4 ra[4]; float4 rb[2];

    auto load_stage = [&](int k0) {
#pragma unroll
        for (int e = 0; e < 4; ++e)
            ra[e] = *reinterpret_cast<const float4*>(&A[(size_t)(m0 + sr + e * 32) * K + k0 + sc]);
#pragma unroll
        for (int e = 0; e < 2; ++e)
            rb[e] = *reinterpret_cast<const float4*>(&W[(size_t)(n0 + sr + e * 32) * K + k0 + sc]);
    };
    auto store_stage = [&](int buf) {
#pragma unroll
        for (int e = 0; e < 4; ++e) {
            __half2* p = reinterpret_cast<__half2*>(&As[buf][(sr + e * 32) * 32 + sc]);
            p[0] = __floats2half2_rn(ra[e].x, ra[e].y);
            p[1] = __floats2half2_rn(ra[e].z, ra[e].w);
        }
#pragma unroll
        for (int e = 0; e < 2; ++e) {
            __half2* p = reinterpret_cast<__half2*>(&Ws[buf][(sr + e * 32) * 32 + sc]);
            p[0] = __floats2half2_rn(rb[e].x, rb[e].y);
            p[1] = __floats2half2_rn(rb[e].z, rb[e].w);
        }
    };

    load_stage(0);
    store_stage(0);
    __syncthreads();

    v8f acc00 = {}, acc01 = {}, acc10 = {}, acc11 = {};
    const int nk = K >> 5;
    for (int ks = 0; ks < nk; ++ks) {
        const int cur = ks & 1;
        if (ks + 1 < nk) load_stage((ks + 1) << 5);   // issue next-slice loads early
        v16h af0 = frag_a_lds(&As[cur][(wy * 32 + l16) * 32], hi);
        v16h af1 = frag_a_lds(&As[cur][(wy * 32 + 16 + l16) * 32], hi);
        v16h bf0 = frag_b_lds(&Ws[cur][(wx * 32 + l16) * 32], hi);
        v16h bf1 = frag_b_lds(&Ws[cur][(wx * 32 + 16 + l16) * 32], hi);
        acc00 = wmma_f16(af0, bf0, acc00);
        acc01 = wmma_f16(af0, bf1, acc01);
        acc10 = wmma_f16(af1, bf0, acc10);
        acc11 = wmma_f16(af1, bf1, acc11);
        if (ks + 1 < nk) {
            store_stage((ks + 1) & 1);
            __syncthreads();   // lowers to s_wait_dscnt + barrier: safe buffer swap
        }
    }

    const int colA = n0 + wx * 32 + l16;
    const int colB = colA + 16;
    const float bA = bias[colA], bB = bias[colB];
#pragma unroll
    for (int r = 0; r < 8; ++r) {
        const int row0 = m0 + wy * 32 + r + 8 * hi;
        const int row1 = row0 + 16;
        float v00 = acc00[r] + bA, v01 = acc01[r] + bB;
        float v10 = acc10[r] + bA, v11 = acc11[r] + bB;
        if (EPI == 1) {
            v00 = fmaxf(v00, 0.f); v01 = fmaxf(v01, 0.f);
            v10 = fmaxf(v10, 0.f); v11 = fmaxf(v11, 0.f);
        }
        if (EPI == 2) {
            if (rowmask[row0]) { v00 = 0.f; v01 = 0.f; }
            if (rowmask[row1]) { v10 = 0.f; v11 = 0.f; }
        }
        store_out(&C[(size_t)row0 * N + colA], v00);
        store_out(&C[(size_t)row0 * N + colB], v01);
        store_out(&C[(size_t)row1 * N + colA], v10);
        store_out(&C[(size_t)row1 * N + colB], v11);
    }
}

// ---------------- flash self-attention ----------------
// qkproj: [BQ,512] cols 0..255 = q proj, 256..511 = k proj. vproj: [BQ,256].
// One wave handles one (b,h,16-row q tile). Block = 8 waves. grid = B*H*(NQ/128)
__global__ __launch_bounds__(256) void attn_kernel(
    const float* __restrict__ qkproj, const float* __restrict__ vproj,
    const unsigned char* __restrict__ qmask, float* __restrict__ out)
{
    __shared__ __half P[8][16 * 32];
    const int lane = threadIdx.x & 31, wv = threadIdx.x >> 5;
    const int bh = blockIdx.x >> 2, qblk = blockIdx.x & 3;
    const int b = bh >> 3, h = bh & 7;
    const int q0 = qblk * 128 + wv * 16;
    const int hi = lane >> 4, l16 = lane & 15;
    const float scale = 0.17677669529663687f;   // 1/sqrt(32)

    const float* qrow = qkproj + (size_t)(b * NQ + q0 + l16) * 512 + h * DHEAD;
    v16h qf = frag_a_g32(qrow, hi);

    v8f o0 = {}, o1 = {};
    float rmax[8], rsum[8];
#pragma unroll
    for (int r = 0; r < 8; ++r) { rmax[r] = -INFINITY; rsum[r] = 0.f; }

    for (int k0 = 0; k0 < NQ; k0 += 32) {
        const float* k0row = qkproj + (size_t)(b * NQ + k0 + l16) * 512 + 256 + h * DHEAD;
        v16h bk0 = frag_b_row_g32(k0row, hi);
        v16h bk1 = frag_b_row_g32(k0row + 16 * 512, hi);
        v8f z = {};
        v8f s0 = wmma_f16(qf, bk0, z);
        v8f s1 = wmma_f16(qf, bk1, z);
        const float m0add = qmask[b * NQ + k0 + l16] ? 0.f : -1e30f;
        const float m1add = qmask[b * NQ + k0 + 16 + l16] ? 0.f : -1e30f;

        float p0[8], p1[8];
#pragma unroll
        for (int r = 0; r < 8; ++r) {
            float a0 = s0[r] * scale + m0add;
            float a1 = s1[r] * scale + m1add;
            float mx = fmaxf(a0, a1);
            for (int d = 1; d < 16; d <<= 1) mx = fmaxf(mx, __shfl_xor(mx, d));
            float nm = fmaxf(rmax[r], mx);
            float corr = __expf(rmax[r] - nm);
            rmax[r] = nm;
            p0[r] = __expf(a0 - nm);
            p1[r] = __expf(a1 - nm);
            float sm = p0[r] + p1[r];
            for (int d = 1; d < 16; d <<= 1) sm += __shfl_xor(sm, d);
            rsum[r] = rsum[r] * corr + sm;
            o0[r] *= corr; o1[r] *= corr;
        }
        // re-layout P (C layout -> A layout) through per-wave LDS (wave-private, in-order DS)
        __half* pl = P[wv];
#pragma unroll
        for (int r = 0; r < 8; ++r) {
            pl[(r + 8 * hi) * 32 + l16]      = __float2half(p0[r]);
            pl[(r + 8 * hi) * 32 + 16 + l16] = __float2half(p1[r]);
        }
        v16h pf = frag_a_lds(&pl[l16 * 32], hi);
        const float* vbase = vproj + (size_t)(b * NQ + k0) * DMODEL + h * DHEAD + l16;
        v16h vf0 = frag_b_strided_g32(vbase, hi, DMODEL);
        v16h vf1 = frag_b_strided_g32(vbase + 16, hi, DMODEL);
        o0 = wmma_f16(pf, vf0, o0);
        o1 = wmma_f16(pf, vf1, o1);
    }
#pragma unroll
    for (int r = 0; r < 8; ++r) {
        int row = q0 + r + 8 * hi;
        float inv = 1.f / rsum[r];
        size_t base = (size_t)(b * NQ + row) * DMODEL + h * DHEAD;
        out[base + l16]      = o0[r] * inv;
        out[base + 16 + l16] = o1[r] * inv;
    }
}

// ---------------- elementwise ----------------
__global__ void add_kernel(const float* __restrict__ a, const float* __restrict__ b,
                           float* __restrict__ o, int n)
{
    int i = blockIdx.x * 256 + threadIdx.x;
    if (i < n) o[i] = a[i] + b[i];
}

// out = LN(a + b) over DF_ features; warp per row, 8 rows/block
template <int DF_>
__global__ __launch_bounds__(256) void ln_kernel(
    const float* __restrict__ a, const float* __restrict__ b,
    const float* __restrict__ g, const float* __restrict__ be,
    float* __restrict__ out)
{
    const int lane = threadIdx.x & 31, wv = threadIdx.x >> 5;
    const size_t row = (size_t)blockIdx.x * 8 + wv;
    const float* ra = a + row * DF_;
    const float* rb = b + row * DF_;
    float x[DF_ / 32];
    float s = 0.f;
#pragma unroll
    for (int i = 0; i < DF_ / 32; ++i) {
        float v = ra[lane + i * 32] + rb[lane + i * 32];
        x[i] = v; s += v;
    }
    for (int d = 1; d < 32; d <<= 1) s += __shfl_xor(s, d);
    float mean = s * (1.f / DF_);
    float vs = 0.f;
#pragma unroll
    for (int i = 0; i < DF_ / 32; ++i) { float dv = x[i] - mean; vs += dv * dv; }
    for (int d = 1; d < 32; d <<= 1) vs += __shfl_xor(vs, d);
    float rstd = rsqrtf(vs * (1.f / DF_) + 1e-5f);
#pragma unroll
    for (int i = 0; i < DF_ / 32; ++i) {
        int j = lane + i * 32;
        out[row * DF_ + j] = (x[i] - mean) * rstd * g[j] + be[j];
    }
}

// cat = LN(concat(tv, ta)) over 512 features
__global__ __launch_bounds__(256) void ln_cat_kernel(
    const float* __restrict__ tv, const float* __restrict__ ta,
    const float* __restrict__ g, const float* __restrict__ be,
    float* __restrict__ out)
{
    const int lane = threadIdx.x & 31, wv = threadIdx.x >> 5;
    const size_t row = (size_t)blockIdx.x * 8 + wv;
    float x[16];
    float s = 0.f;
#pragma unroll
    for (int i = 0; i < 16; ++i) {
        int j = lane + i * 32;
        float v = (j < 256) ? tv[row * 256 + j] : ta[row * 256 + j - 256];
        x[i] = v; s += v;
    }
    for (int d = 1; d < 32; d <<= 1) s += __shfl_xor(s, d);
    float mean = s * (1.f / 512.f);
    float vs = 0.f;
#pragma unroll
    for (int i = 0; i < 16; ++i) { float dv = x[i] - mean; vs += dv * dv; }
    for (int d = 1; d < 32; d <<= 1) vs += __shfl_xor(vs, d);
    float rstd = rsqrtf(vs * (1.f / 512.f) + 1e-5f);
#pragma unroll
    for (int i = 0; i < 16; ++i) {
        int j = lane + i * 32;
        out[row * 512 + j] = (x[i] - mean) * rstd * g[j] + be[j];
    }
}

// ---------------- MSDA prep: softmax(attn), loc = ref + off/T ----------------
// One thread per (b,q,h). offsets/attn are shared between video & audio.
__global__ void msda_prep_kernel(
    const float* __restrict__ offw, const float* __restrict__ attnw,
    const float* __restrict__ refv, const float* __restrict__ refa,
    float* __restrict__ vattn, float* __restrict__ aattn,
    float* __restrict__ vloc, float* __restrict__ aloc)
{
    int t = blockIdx.x * 256 + threadIdx.x;         // < BQ*NHEAD
    int h = t & 7, bq = t >> 3;
    const float* ow = offw + (size_t)bq * 128 + h * 16;
    const float* aw = attnw + (size_t)bq * 128 + h * 16;
    const float Tsh[4] = { 8192.f, 4096.f, 2048.f, 1024.f };
    float av[16], mx = -1e38f;
#pragma unroll
    for (int lp = 0; lp < 16; ++lp) { av[lp] = aw[lp]; mx = fmaxf(mx, av[lp]); }
    float s = 0.f;
#pragma unroll
    for (int lp = 0; lp < 16; ++lp) { av[lp] = __expf(av[lp] - mx); s += av[lp]; }
    float inv = 1.f / s;
#pragma unroll
    for (int lp = 0; lp < 16; ++lp) {
        int l = lp >> 2;
        size_t oidx = (size_t)t * 16 + lp;
        float at = av[lp] * inv;
        vattn[oidx] = at; aattn[oidx] = at;
        float off = ow[lp] / Tsh[l];
        vloc[oidx] = refv[(size_t)bq * 4 + l] + off;
        aloc[oidx] = refa[(size_t)bq * 4 + l] + off;
    }
}

// ---------------- MSDA sampling: warp per (b,q,h), lane = dh ----------------
__global__ __launch_bounds__(256) void msda_sample_kernel(
    const __half* __restrict__ val, const float* __restrict__ attn,
    const float* __restrict__ loc, float* __restrict__ out)
{
    const int lane = threadIdx.x & 31;
    const int g = (blockIdx.x * 256 + threadIdx.x) >> 5;  // < BQ*NHEAD
    const int h = g & 7, bq = g >> 3, b = bq >> 9;
    const int   Tsh[4] = { 8192, 4096, 2048, 1024 };
    const int   lsi[4] = { 0, 8192, 12288, 14336 };
    const float* at = attn + (size_t)g * 16;
    const float* lc = loc + (size_t)g * 16;
    float acc = 0.f;
#pragma unroll
    for (int lp = 0; lp < 16; ++lp) {
        int l = lp >> 2;
        int T = Tsh[l];
        float a = at[lp];
        float x = lc[lp] * (float)T - 0.5f;
        float x0 = floorf(x);
        float wg = x - x0;
        int i0 = (int)x0, i1 = i0 + 1;
        float m0 = (i0 >= 0 && i0 < T) ? 1.f : 0.f;
        float m1 = (i1 >= 0 && i1 < T) ? 1.f : 0.f;
        int c0 = i0 < 0 ? 0 : (i0 > T - 1 ? T - 1 : i0);
        int c1 = i1 < 0 ? 0 : (i1 > T - 1 ? T - 1 : i1);
        const __half* vb = val + ((size_t)(b * STOT + lsi[l])) * DMODEL + h * DHEAD + lane;
        float g0 = __half2float(vb[(size_t)c0 * DMODEL]);
        float g1 = __half2float(vb[(size_t)c1 * DMODEL]);
        acc += a * (m0 * (1.f - wg) * g0 + m1 * wg * g1);
    }
    out[(size_t)bq * DMODEL + h * DHEAD + lane] = acc;
}

// ---------------- launch ----------------
extern "C" void kernel_launch(void* const* d_in, const int* in_sizes, int n_in,
                              void* d_out, int out_size, void* d_ws, size_t ws_size,
                              hipStream_t stream)
{
    (void)in_sizes; (void)n_in; (void)out_size; (void)ws_size;
    const float* tgt  = (const float*)d_in[0];
    const float* qpos = (const float*)d_in[1];
    const float* refv = (const float*)d_in[2];
    const float* refa = (const float*)d_in[3];
    const unsigned char* qmask = (const unsigned char*)d_in[4];
    const float* vsrc = (const float*)d_in[5];
    const unsigned char* vpad = (const unsigned char*)d_in[8];
    const float* asrc = (const float*)d_in[9];
    const unsigned char* apad = (const unsigned char*)d_in[12];
    const float* sa_in_w  = (const float*)d_in[13];
    const float* sa_in_b  = (const float*)d_in[14];
    const float* sa_out_w = (const float*)d_in[15];
    const float* sa_out_b = (const float*)d_in[16];
    const float* ca_off_w = (const float*)d_in[17];
    const float* ca_off_b = (const float*)d_in[18];
    const float* ca_attn_w = (const float*)d_in[19];
    const float* ca_attn_b = (const float*)d_in[20];
    const float* ca_val_w = (const float*)d_in[21];
    const float* ca_val_b = (const float*)d_in[22];
    const float* ca_out_w = (const float*)d_in[23];
    const float* ca_out_b = (const float*)d_in[24];
    const float* n1_w = (const float*)d_in[25];
    const float* n1_b = (const float*)d_in[26];
    const float* n2_w = (const float*)d_in[27];
    const float* n2_b = (const float*)d_in[28];
    const float* n3_w = (const float*)d_in[29];
    const float* n3_b = (const float*)d_in[30];
    const float* n4_w = (const float*)d_in[31];
    const float* n4_b = (const float*)d_in[32];
    const float* lin1_w = (const float*)d_in[33];
    const float* lin1_b = (const float*)d_in[34];
    const float* lin2_w = (const float*)d_in[35];
    const float* lin2_b = (const float*)d_in[36];
    const float* lin3_w = (const float*)d_in[37];
    const float* lin3_b = (const float*)d_in[38];

    float* outp = (float*)d_out;
    // output slots (reference returns: out, tv, ta, vloc, vattn, aloc, aattn)
    float* o_out   = outp;
    float* o_tv    = outp + 2097152;
    float* o_ta    = outp + 4194304;
    float* o_vloc  = outp + 6291456;
    float* o_vattn = outp + 7340032;
    float* o_aloc  = outp + 8388608;
    float* o_aattn = outp + 9437184;

    // workspace layout (floats)
    float* w = (float*)d_ws;
    float* qk     = w;               w += 2097152;
    float* t      = w;               w += 2097152;
    float* qv     = w;               w += 2097152;
    float* qkproj = w;               w += 4194304;
    float* vproj  = w;               w += 2097152;
    float* attno  = w;               w += 2097152;
    float* t2     = w;               w += 2097152;
    float* offw   = w;               w += 1048576;
    float* attnw  = w;               w += 1048576;
    float* samp   = w;               w += 2097152;
    float* cv     = w;               w += 2097152;
    float* cat    = w;               w += 4194304;
    float* xbuf   = w;               w += 2097152;
    float* ffbuf  = w;               w += 2097152;
    float* h1     = w;               w += 8388608;
    __half* val   = (__half*)w;      // BS*256 halves = 31457280 floats worth / 2

    const dim3 blk(256);
    // 1. qk = tgt + query_pos
    add_kernel<<<8192, blk, 0, stream>>>(tgt, qpos, qk, BQ * DMODEL);
    // 2. q|k projection: [BQ,512] = qk @ sa_in_w[0:512]^T + b
    gemm_kernel<0, float><<<dim3(8, 64), blk, 0, stream>>>(qk, sa_in_w, sa_in_b, qkproj, BQ, 512, 256, nullptr);
    // 3. v projection: tgt @ wv^T
    gemm_kernel<0, float><<<dim3(4, 64), blk, 0, stream>>>(tgt, sa_in_w + 512 * 256, sa_in_b + 512, vproj, BQ, 256, 256, nullptr);
    // 4. flash self-attention
    attn_kernel<<<BATCH * NHEAD * 4, blk, 0, stream>>>(qkproj, vproj, qmask, attno);
    // 5. output projection
    gemm_kernel<0, float><<<dim3(4, 64), blk, 0, stream>>>(attno, sa_out_w, sa_out_b, t2, BQ, 256, 256, nullptr);
    // 6. t = LN(tgt + t2)
    ln_kernel<256><<<1024, blk, 0, stream>>>(tgt, t2, n2_w, n2_b, t);
    // 7. qv = t + query_pos
    add_kernel<<<8192, blk, 0, stream>>>(t, qpos, qv, BQ * DMODEL);
    // 8/9. offsets + attention weights (shared by both modalities)
    gemm_kernel<0, float><<<dim3(2, 64), blk, 0, stream>>>(qv, ca_off_w, ca_off_b, offw, BQ, 128, 256, nullptr);
    gemm_kernel<0, float><<<dim3(2, 64), blk, 0, stream>>>(qv, ca_attn_w, ca_attn_b, attnw, BQ, 128, 256, nullptr);
    // 10. softmax + sampling locations -> outputs
    msda_prep_kernel<<<256, blk, 0, stream>>>(offw, attnw, refv, refa, o_vattn, o_aattn, o_vloc, o_aloc);
    // ---- video branch ----
    gemm_kernel<2, __half><<<dim3(4, 1920), blk, 0, stream>>>(vsrc, ca_val_w, ca_val_b, val, BS, 256, 256, vpad);
    msda_sample_kernel<<<8192, blk, 0, stream>>>(val, o_vattn, o_vloc, samp);
    gemm_kernel<0, float><<<dim3(4, 64), blk, 0, stream>>>(samp, ca_out_w, ca_out_b, cv, BQ, 256, 256, nullptr);
    ln_kernel<256><<<1024, blk, 0, stream>>>(t, cv, n1_w, n1_b, o_tv);
    // ---- audio branch (reuse val/samp/cv) ----
    gemm_kernel<2, __half><<<dim3(4, 1920), blk, 0, stream>>>(asrc, ca_val_w, ca_val_b, val, BS, 256, 256, apad);
    msda_sample_kernel<<<8192, blk, 0, stream>>>(val, o_aattn, o_aloc, samp);
    gemm_kernel<0, float><<<dim3(4, 64), blk, 0, stream>>>(samp, ca_out_w, ca_out_b, cv, BQ, 256, 256, nullptr);
    ln_kernel<256><<<1024, blk, 0, stream>>>(t, cv, n1_w, n1_b, o_ta);
    // ---- fusion + FFN ----
    ln_cat_kernel<<<1024, blk, 0, stream>>>(o_tv, o_ta, n4_w, n4_b, cat);
    gemm_kernel<1, float><<<dim3(4, 64), blk, 0, stream>>>(cat, lin3_w, lin3_b, xbuf, BQ, 256, 512, nullptr);
    gemm_kernel<1, float><<<dim3(16, 64), blk, 0, stream>>>(xbuf, lin1_w, lin1_b, h1, BQ, 1024, 256, nullptr);
    gemm_kernel<0, float><<<dim3(4, 64), blk, 0, stream>>>(h1, lin2_w, lin2_b, ffbuf, BQ, 256, 1024, nullptr);
    ln_kernel<256><<<1024, blk, 0, stream>>>(xbuf, ffbuf, n3_w, n3_b, o_out);
}
